// Conv2d_71116068487874
// MI455X (gfx1250) — matrix operand order repaired and checked by reading, not compiled, not run on hardware
//
#include <hip/hip_runtime.h>

typedef __attribute__((ext_vector_type(8)))  _Float16 v8h;
typedef __attribute__((ext_vector_type(16))) _Float16 v16h;
typedef __attribute__((ext_vector_type(8)))  float    v8f;
typedef __attribute__((ext_vector_type(4)))  unsigned v4u;

// Problem constants
static constexpr int BATCH = 32;
static constexpr int CIN   = 128;
static constexpr int HH    = 56;
static constexpr int WW    = 56;
static constexpr int COUT  = 256;
static constexpr int HWSZ  = HH * WW;        // 3136
static constexpr int KK    = CIN * 9;        // 1152 (GEMM K)
static constexpr int NTOT  = BATCH * HWSZ;   // 100352 (GEMM N)

// Tiling
static constexpr int BM  = 128;
static constexpr int BN  = 128;
static constexpr int BK  = 32;
static constexpr int LDH = BK + 8;           // 40 halves/row: 16B-aligned runs, staggered banks

// ---------------------------------------------------------------------------
// Pre-kernel: split fp32 weights into hi/lo fp16 AND permute K to
// position-major order:  kperm = r*128 + ci   (r = ky*3+kx)
// so the main kernel's im2col decode is loop-invariant per K-chunk.
// ---------------------------------------------------------------------------
__global__ void split_permute_weights_kernel(const float* __restrict__ w,
                                             _Float16* __restrict__ wh,
                                             _Float16* __restrict__ wl, int n)
{
    int i = blockIdx.x * 256 + threadIdx.x;   // destination (permuted) index
    if (i < n) {
        const int m  = i / KK;
        const int t  = i - m * KK;            // r*128 + ci
        const int r  = t >> 7;                // kernel position 0..8
        const int ci = t & 127;
        float v = w[m * KK + ci * 9 + r];
        _Float16 h = (_Float16)v;                 // RNE to fp16
        _Float16 l = (_Float16)(v - (float)h);    // exact residual, then RNE
        wh[i] = h;
        wl[i] = l;
    }
}

// ---------------------------------------------------------------------------
// Main kernel: implicit-GEMM conv via fp16x3 WMMA (fp32-accurate)
// ---------------------------------------------------------------------------
__global__ __launch_bounds__(256, 1)
void conv2d_wmma_f16x3(const float* __restrict__ x,
                       const _Float16* __restrict__ wh,
                       const _Float16* __restrict__ wl,
                       const float* __restrict__ bias,
                       float* __restrict__ out)
{
    __shared__ __align__(16) _Float16 Ah[BM * LDH];
    __shared__ __align__(16) _Float16 Al[BM * LDH];
    __shared__ __align__(16) _Float16 Bh[BN * LDH];
    __shared__ __align__(16) _Float16 Bl[BN * LDH];

    const int tid  = threadIdx.x;          // 0..255
    const int lane = tid & 31;
    const int wave = tid >> 5;             // 0..7
    const int wm   = wave & 1;             // M half  (64 rows each)
    const int wn   = wave >> 1;            // N quarter (32 cols each)
    const int l15  = lane & 15;
    const int lhi  = (lane >> 4) & 1;

    const int m0 = blockIdx.y * BM;
    const int n0 = blockIdx.x * BN;

    // ---- per-thread im2col column decode (fixed over K loop) ----
    const int nb    = tid & 127;           // B-tile column
    const int gn    = n0 + nb;
    const int b_img = gn / HWSZ;
    const int p_img = gn - b_img * HWSZ;
    const int oy    = p_img / WW;
    const int ox    = p_img - oy * WW;
    const int bhalf = tid >> 7;            // 0..1 : which 16-k half of the row

    // ---- A loader mapping: each thread owns one contiguous 16-half run ----
    const int a_m = tid >> 1;              // 0..127
    const int a_q = (tid & 1) * 16;        // 0 or 16 (half offset within row)

    unsigned* Bh32 = (unsigned*)Bh;
    unsigned* Bl32 = (unsigned*)Bl;

    v8f acc[4][2];
    #pragma unroll
    for (int i = 0; i < 4; ++i)
        #pragma unroll
        for (int j = 0; j < 2; ++j)
            acc[i][j] = (v8f){0.f, 0.f, 0.f, 0.f, 0.f, 0.f, 0.f, 0.f};

    for (int k0 = 0; k0 < KK; k0 += BK) {
        // ---- Stage A tile (pre-split, pre-permuted weights): 128 m x 32 k ----
        // 4x global_load_b128 + 4x ds_store_b128 per thread.
        {
            const _Float16* gh = wh + (m0 + a_m) * KK + k0 + a_q;
            const _Float16* gl = wl + (m0 + a_m) * KK + k0 + a_q;
            v8h h0 = *(const v8h*)(gh);
            v8h h1 = *(const v8h*)(gh + 8);
            v8h l0 = *(const v8h*)(gl);
            v8h l1 = *(const v8h*)(gl + 8);
            *(v8h*)&Ah[a_m * LDH + a_q]     = h0;
            *(v8h*)&Ah[a_m * LDH + a_q + 8] = h1;
            *(v8h*)&Al[a_m * LDH + a_q]     = l0;
            *(v8h*)&Al[a_m * LDH + a_q + 8] = l1;
        }

        // ---- Stage B tile (im2col, position-major K): 128 n x 32 k ----
        // Whole chunk lies inside one kernel position r -> decode ONCE.
        // Each thread covers 16 contiguous k (8 pairs) of its row -> b128 stores.
        {
            const int r   = k0 >> 7;                   // kernel position (uniform)
            const int ci0 = (k0 & 127) + bhalf * 16;   // this thread's first ci
            const int ky  = r / 3;
            const int kx  = r - 3 * ky;
            const int iy  = oy + ky - 1;
            const int ix  = ox + kx - 1;
            const bool valid = ((unsigned)iy < (unsigned)HH) &&
                               ((unsigned)ix < (unsigned)WW);
            const float* xp = x + (((long)b_img * CIN + ci0) * HH + iy) * WW + ix;

            unsigned uh[8], ul[8];
            #pragma unroll
            for (int j = 0; j < 8; ++j) {
                float v0 = 0.0f, v1 = 0.0f;
                if (valid) {
                    v0 = xp[(2 * j)     * HWSZ];
                    v1 = xp[(2 * j + 1) * HWSZ];
                }
                union { _Float16 h[2]; unsigned u; } ph, pl;
                _Float16 h0 = (_Float16)v0;
                _Float16 h1 = (_Float16)v1;
                ph.h[0] = h0;  ph.h[1] = h1;
                pl.h[0] = (_Float16)(v0 - (float)h0);
                pl.h[1] = (_Float16)(v1 - (float)h1);
                uh[j] = ph.u;  ul[j] = pl.u;
            }
            unsigned* dh = &Bh32[nb * (LDH / 2) + bhalf * 8];
            unsigned* dl = &Bl32[nb * (LDH / 2) + bhalf * 8];
            *(v4u*)(dh)     = (v4u){uh[0], uh[1], uh[2], uh[3]};
            *(v4u*)(dh + 4) = (v4u){uh[4], uh[5], uh[6], uh[7]};
            *(v4u*)(dl)     = (v4u){ul[0], ul[1], ul[2], ul[3]};
            *(v4u*)(dl + 4) = (v4u){ul[4], ul[5], ul[6], ul[7]};
        }
        __syncthreads();

        // ---- Fragments ----
        // A (16-bit, 16x32): lane m = l15; contiguous K run [16*lhi, +16)
        v16h fAh[4], fAl[4];
        #pragma unroll
        for (int t = 0; t < 4; ++t) {
            const int m = wm * 64 + t * 16 + l15;
            v8h a0 = *(const v8h*)&Ah[m * LDH + 16 * lhi];
            v8h a1 = *(const v8h*)&Ah[m * LDH + 16 * lhi + 8];
            v8h b0 = *(const v8h*)&Al[m * LDH + 16 * lhi];
            v8h b1 = *(const v8h*)&Al[m * LDH + 16 * lhi + 8];
            fAh[t] = __builtin_shufflevector(a0, a1, 0,1,2,3,4,5,6,7,8,9,10,11,12,13,14,15);
            fAl[t] = __builtin_shufflevector(b0, b1, 0,1,2,3,4,5,6,7,8,9,10,11,12,13,14,15);
        }
        // B (16-bit, 32x16): lane n = l15; contiguous K run [16*lhi, +16)
        v16h fBh[2], fBl[2];
        #pragma unroll
        for (int t = 0; t < 2; ++t) {
            const int n = wn * 32 + t * 16 + l15;
            const _Float16* rh = &Bh[n * LDH + 16 * lhi];
            const _Float16* rl = &Bl[n * LDH + 16 * lhi];
            v8h a0 = *(const v8h*)(rh), a1 = *(const v8h*)(rh + 8);
            v8h b0 = *(const v8h*)(rl), b1 = *(const v8h*)(rl + 8);
            fBh[t] = __builtin_shufflevector(a0, a1, 0,1,2,3,4,5,6,7,8,9,10,11,12,13,14,15);
            fBl[t] = __builtin_shufflevector(b0, b1, 0,1,2,3,4,5,6,7,8,9,10,11,12,13,14,15);
        }

        // ---- fp16x3: ah*bh + ah*bl + al*bh, fp32 accumulate ----
        #pragma unroll
        for (int tm = 0; tm < 4; ++tm) {
            #pragma unroll
            for (int tn = 0; tn < 2; ++tn) {
                acc[tm][tn] = __builtin_amdgcn_wmma_f32_16x16x32_f16(
                    false, fAh[tm], false, fBh[tn], (short)0, acc[tm][tn], false, false);
                acc[tm][tn] = __builtin_amdgcn_wmma_f32_16x16x32_f16(
                    false, fAh[tm], false, fBl[tn], (short)0, acc[tm][tn], false, false);
                acc[tm][tn] = __builtin_amdgcn_wmma_f32_16x16x32_f16(
                    false, fAl[tm], false, fBh[tn], (short)0, acc[tm][tn], false, false);
            }
        }
        __syncthreads();
    }

    // ---- Epilogue: bias + scatter to NCHW ----
    // C/D: VGPR r -> M = r + 8*lhi, N = l15 within each 16x16 tile
    #pragma unroll
    for (int tn = 0; tn < 2; ++tn) {
        const int gcol = n0 + wn * 32 + tn * 16 + l15;
        const int b    = gcol / HWSZ;
        const int p    = gcol - b * HWSZ;
        const long base = (long)b * COUT * HWSZ + p;
        #pragma unroll
        for (int tm = 0; tm < 4; ++tm) {
            #pragma unroll
            for (int r = 0; r < 8; ++r) {
                const int m = m0 + wm * 64 + tm * 16 + r + 8 * lhi;
                out[base + (long)m * HWSZ] = acc[tm][tn][r] + bias[m];
            }
        }
    }
}

extern "C" void kernel_launch(void* const* d_in, const int* in_sizes, int n_in,
                              void* d_out, int out_size, void* d_ws, size_t ws_size,
                              hipStream_t stream)
{
    const float* x    = (const float*)d_in[0];   // (32,128,56,56)
    const float* w    = (const float*)d_in[1];   // (256,128,3,3)
    const float* bias = (const float*)d_in[2];   // (256,)
    float* out = (float*)d_out;                  // (32,256,56,56)

    const int nw = COUT * KK;                    // 294912 weight elements
    _Float16* wh = (_Float16*)d_ws;              // hi halves (0.59 MB)
    _Float16* wl = wh + nw;                      // lo halves (0.59 MB)

    split_permute_weights_kernel<<<(nw + 255) / 256, 256, 0, stream>>>(w, wh, wl, nw);

    dim3 grid(NTOT / BN, COUT / BM);             // 784 x 2
    dim3 block(256);
    conv2d_wmma_f16x3<<<grid, block, 0, stream>>>(x, wh, wl, bias, out);
}
